// SpatialMultiHeadAttentionMIL_33947421508228
// MI455X (gfx1250) — matrix operand order repaired and checked by reading, not compiled, not run on hardware
//
#include <hip/hip_runtime.h>

#define N_TOK 4096
#define DIM_  384
#define NH    6
#define DH    64
#define MAXD  20
#define KDIM  384

#define QT 64      // query rows per attention block
#define KT 64      // key tile
#define WAVES 4
#define PADH 72    // padded LDS row length in halves (72*2=144B, conflict-free for 16-row strides)

typedef __attribute__((ext_vector_type(16))) _Float16 v16h;
typedef __attribute__((ext_vector_type(8)))  _Float16 v8h;
typedef __attribute__((ext_vector_type(8)))  float    v8f;
typedef __attribute__((ext_vector_type(4)))  int      v4i;

typedef __attribute__((address_space(1))) v4i as1_v4i;
typedef __attribute__((address_space(3))) v4i as3_v4i;

union V16U { v16h v; v8h h[2]; };

#if __has_builtin(__builtin_amdgcn_global_load_async_to_lds_b128)
#define HAVE_ASYNC 1
#else
#define HAVE_ASYNC 0
#endif

#if HAVE_ASYNC
#if __has_builtin(__builtin_amdgcn_s_wait_asynccnt)
#define ASYNC_WAIT() __builtin_amdgcn_s_wait_asynccnt(0)
#else
#define ASYNC_WAIT() asm volatile("s_wait_asynccnt 0" ::: "memory")
#endif
#else
#define ASYNC_WAIT() ((void)0)
#endif

__device__ __forceinline__ void cp16(_Float16* l, const _Float16* g) {
#if HAVE_ASYNC
  __builtin_amdgcn_global_load_async_to_lds_b128(
      (as1_v4i*)(void*)g, (as3_v4i*)(void*)l, 0, 0);
#else
  *(v8h*)l = *(const v8h*)g;
#endif
}

// ---------------- conversion kernels ----------------
__global__ void cvt_f16_kernel(const float* __restrict__ src, _Float16* __restrict__ dst, int n) {
  int i = blockIdx.x * blockDim.x + threadIdx.x;
  if (i < n) dst[i] = (_Float16)src[i];
}

__global__ void cvt_dist4_kernel(const int4* __restrict__ d, unsigned int* __restrict__ o, int n4) {
  int i = blockIdx.x * blockDim.x + threadIdx.x;
  if (i < n4) {
    int4 v = d[i];
    int a = v.x < 0 ? 0 : (v.x > MAXD ? MAXD : v.x);
    int b = v.y < 0 ? 0 : (v.y > MAXD ? MAXD : v.y);
    int c = v.z < 0 ? 0 : (v.z > MAXD ? MAXD : v.z);
    int e = v.w < 0 ? 0 : (v.w > MAXD ? MAXD : v.w);
    o[i] = (unsigned)a | ((unsigned)b << 8) | ((unsigned)c << 16) | ((unsigned)e << 24);
  }
}

// ---------------- QKV projection GEMM (WMMA f16 -> f32 acc) ----------------
// out o = s*384 + h*64 + d ; q/k stored [h][n][64], v stored transposed [h][64][n]
__global__ __launch_bounds__(256) void qkv_gemm_kernel(
    const _Float16* __restrict__ xh, const _Float16* __restrict__ wh,
    const float* __restrict__ qb,
    _Float16* __restrict__ qh, _Float16* __restrict__ kh, _Float16* __restrict__ vth)
{
  const int tid = threadIdx.x;
  const int wave = tid >> 5, lane = tid & 31;
  const int lhalf = lane >> 4, l16 = lane & 15;
  const int rbase = blockIdx.x * 128 + wave * 16;   // 16 output rows per wave
  const int cbase = blockIdx.y * 32;                // 32 output cols per wave

  v8f cf[2] = {};
  const _Float16* xr = xh + (size_t)(rbase + l16) * KDIM;
#pragma unroll
  for (int c = 0; c < KDIM / 32; ++c) {
    V16U a;
    a.h[0] = *(const v8h*)(xr + 32 * c + 16 * lhalf);       // K = 16*hi + 0..7
    a.h[1] = *(const v8h*)(xr + 32 * c + 16 * lhalf + 8);   // K = 16*hi + 8..15
#pragma unroll
    for (int g = 0; g < 2; ++g) {
      const _Float16* wr = wh + (size_t)(cbase + 16 * g + l16) * KDIM + 32 * c + 16 * lhalf;
      V16U b;
      b.h[0] = *(const v8h*)wr;          // K = 16*hi + 0..7
      b.h[1] = *(const v8h*)(wr + 8);    // K = 16*hi + 8..15
      cf[g] = __builtin_amdgcn_wmma_f32_16x16x32_f16(false, a.v, false, b.v,
                                                     (short)0, cf[g], false, false);
    }
  }
#pragma unroll
  for (int g = 0; g < 2; ++g) {
    int o = cbase + 16 * g + l16;
    int s = o / DIM_;
    int rem = o - s * DIM_;
    int h = rem >> 6, d = rem & 63;
    float bias = qb[o];
#pragma unroll
    for (int i = 0; i < 8; ++i) {
      int n = rbase + i + 8 * lhalf;
      _Float16 hv = (_Float16)(cf[g][i] + bias);
      if (s == 0)      qh [((size_t)h * N_TOK + n) * DH + d] = hv;
      else if (s == 1) kh [((size_t)h * N_TOK + n) * DH + d] = hv;
      else             vth[((size_t)h * DH + d) * N_TOK + n] = hv;
    }
  }
}

// ---------------- flash attention with RPE bias ----------------
__global__ __launch_bounds__(128) void flash_attn_kernel(
    const _Float16* __restrict__ qh, const _Float16* __restrict__ kh,
    const _Float16* __restrict__ vth, const unsigned char* __restrict__ du8,
    const float* __restrict__ rpe, float* __restrict__ out)
{
  __shared__ _Float16 klds[KT][PADH];          // [key][d]
  __shared__ _Float16 vlds[DH][PADH];          // [d][key]  (v transposed tile)
  __shared__ _Float16 plds[WAVES][16][PADH];   // per-wave P bounce [row][key]
  __shared__ float rpeS[MAXD + 1];

  const int h   = blockIdx.y;
  const int tq  = blockIdx.x * QT;
  const int tid = threadIdx.x;
  const int wave = tid >> 5, lane = tid & 31;
  const int lhalf = lane >> 4, l16 = lane & 15;
  const float LOG2E = 1.4426950408889634f;

  if (tid <= MAXD) rpeS[tid] = rpe[tid * NH + h];

  // Q fragments (16 rows per wave), resident for whole kernel
  const int qrow = tq + wave * 16 + l16;
  const _Float16* qbase = qh + ((size_t)h * N_TOK + qrow) * DH;
  V16U qf[2];
#pragma unroll
  for (int c = 0; c < 2; ++c) {
    qf[c].h[0] = *(const v8h*)(qbase + 32 * c + 16 * lhalf);
    qf[c].h[1] = *(const v8h*)(qbase + 32 * c + 16 * lhalf + 8);
  }

  v8f o4[4] = {};                   // O accumulators: 16 rows x 64 cols
  float m2[8], lsum[8];
#pragma unroll
  for (int i = 0; i < 8; ++i) { m2[i] = -1.0e30f; lsum[i] = 0.0f; }

  const unsigned char* dq = du8 + (size_t)(tq + wave * 16) * N_TOK;

  for (int kt = 0; kt < N_TOK; kt += KT) {
    __syncthreads();   // previous tile fully consumed
    {
      int row  = tid >> 1;          // 0..63
      int half = tid & 1;           // 64B each
      const _Float16* gk = kh  + ((size_t)h * N_TOK + kt + row) * DH + half * 32;
      const _Float16* gv = vth + ((size_t)h * DH + row) * N_TOK + kt + half * 32;
      _Float16* lk = &klds[row][half * 32];
      _Float16* lv = &vlds[row][half * 32];
#pragma unroll
      for (int s = 0; s < 4; ++s) {
        cp16(lk + 8 * s, gk + 8 * s);
        cp16(lv + 8 * s, gv + 8 * s);
      }
    }
    ASYNC_WAIT();
    __syncthreads();   // tile visible to all waves

    // S = Q K^T : 4 key groups of 16, 2 d-chunks of 32
    v8f s4[4];
#pragma unroll
    for (int f = 0; f < 4; ++f) {
      v8f acc = {};
#pragma unroll
      for (int c = 0; c < 2; ++c) {
        const _Float16* kp = &klds[f * 16 + l16][c * 32 + 16 * lhalf];
        V16U b; b.h[0] = *(const v8h*)kp; b.h[1] = *(const v8h*)(kp + 8);
        acc = __builtin_amdgcn_wmma_f32_16x16x32_f16(false, qf[c].v, false, b.v,
                                                     (short)0, acc, false, false);
      }
      s4[f] = acc;
    }

    // add relative-position bias (u8 distance -> LDS table)
#pragma unroll
    for (int f = 0; f < 4; ++f) {
      int kcol = kt + f * 16 + l16;
#pragma unroll
      for (int i = 0; i < 8; ++i) {
        unsigned char dv = dq[(size_t)(i + 8 * lhalf) * N_TOK + kcol];
        s4[f][i] += rpeS[dv];
      }
    }

    // row max across 4 frags then across the 16 lanes holding each row
    float rmax[8];
#pragma unroll
    for (int i = 0; i < 8; ++i)
      rmax[i] = fmaxf(fmaxf(s4[0][i], s4[1][i]), fmaxf(s4[2][i], s4[3][i]));
#pragma unroll
    for (int msk = 1; msk <= 8; msk <<= 1)
#pragma unroll
      for (int i = 0; i < 8; ++i)
        rmax[i] = fmaxf(rmax[i], __shfl_xor(rmax[i], msk, 32));

    // online softmax (exp2 domain)
    float scale[8];
#pragma unroll
    for (int i = 0; i < 8; ++i) {
      float mn = fmaxf(m2[i], rmax[i] * LOG2E);
      scale[i] = __builtin_amdgcn_exp2f(m2[i] - mn);
      m2[i] = mn;
      lsum[i] *= scale[i];
    }
#pragma unroll
    for (int g = 0; g < 4; ++g)
#pragma unroll
      for (int i = 0; i < 8; ++i)
        o4[g][i] *= scale[i];

    float psum[8] = {0.f, 0.f, 0.f, 0.f, 0.f, 0.f, 0.f, 0.f};
#pragma unroll
    for (int f = 0; f < 4; ++f)
#pragma unroll
      for (int i = 0; i < 8; ++i) {
        float e = __builtin_amdgcn_exp2f(s4[f][i] * LOG2E - m2[i]);
        psum[i] += e;
        plds[wave][i + 8 * lhalf][f * 16 + l16] = (_Float16)e;
      }
#pragma unroll
    for (int msk = 1; msk <= 8; msk <<= 1)
#pragma unroll
      for (int i = 0; i < 8; ++i)
        psum[i] += __shfl_xor(psum[i], msk, 32);
#pragma unroll
    for (int i = 0; i < 8; ++i) lsum[i] += psum[i];

    // wave-private LDS bounce: D-layout -> A-layout (LDS is in-order per wave)
    asm volatile("s_wait_dscnt 0" ::: "memory");

    V16U pa[2];
#pragma unroll
    for (int c = 0; c < 2; ++c) {
      const _Float16* pp = &plds[wave][l16][c * 32 + 16 * lhalf];
      pa[c].h[0] = *(const v8h*)pp;
      pa[c].h[1] = *(const v8h*)(pp + 8);
    }
    // O += P * V : 4 d-column groups, 2 key-chunks of 32
#pragma unroll
    for (int g = 0; g < 4; ++g)
#pragma unroll
      for (int c = 0; c < 2; ++c) {
        const _Float16* vp = &vlds[g * 16 + l16][c * 32 + 16 * lhalf];
        V16U b; b.h[0] = *(const v8h*)vp; b.h[1] = *(const v8h*)(vp + 8);
        o4[g] = __builtin_amdgcn_wmma_f32_16x16x32_f16(false, pa[c].v, false, b.v,
                                                       (short)0, o4[g], false, false);
      }
  }

  // epilogue: normalize and store f32 output [n][h*64+d]
  float invl[8];
#pragma unroll
  for (int i = 0; i < 8; ++i) invl[i] = 1.0f / lsum[i];
  float* ob = out + (size_t)(tq + wave * 16) * DIM_ + h * DH;
#pragma unroll
  for (int g = 0; g < 4; ++g)
#pragma unroll
    for (int i = 0; i < 8; ++i)
      ob[(size_t)(i + 8 * lhalf) * DIM_ + g * 16 + l16] = o4[g][i] * invl[i];
}

// ---------------- launch ----------------
extern "C" void kernel_launch(void* const* d_in, const int* in_sizes, int n_in,
                              void* d_out, int out_size, void* d_ws, size_t ws_size,
                              hipStream_t stream) {
  const float* x      = (const float*)d_in[0];
  const int*   dist   = (const int*)d_in[1];
  const float* qkv_w  = (const float*)d_in[2];
  const float* qkv_b  = (const float*)d_in[3];
  const float* rpe    = (const float*)d_in[4];
  float* out = (float*)d_out;

  char* base = (char*)d_ws;
  size_t off = 0;
  auto take = [&](size_t bytes) -> void* {
    void* p = base + off;
    off += (bytes + 255) & ~(size_t)255;
    return p;
  };
  _Float16* xh  = (_Float16*)take((size_t)N_TOK * KDIM * 2);
  _Float16* wh  = (_Float16*)take((size_t)3 * DIM_ * KDIM * 2);
  _Float16* qh  = (_Float16*)take((size_t)NH * N_TOK * DH * 2);
  _Float16* kh  = (_Float16*)take((size_t)NH * N_TOK * DH * 2);
  _Float16* vth = (_Float16*)take((size_t)NH * N_TOK * DH * 2);
  unsigned char* du8 = (unsigned char*)take((size_t)N_TOK * N_TOK);

  int nx = N_TOK * KDIM;        // 1,572,864
  int nw = 3 * DIM_ * KDIM;     //   442,368
  cvt_f16_kernel<<<(nx + 255) / 256, 256, 0, stream>>>(x, xh, nx);
  cvt_f16_kernel<<<(nw + 255) / 256, 256, 0, stream>>>(qkv_w, wh, nw);

  int nd4 = (N_TOK * N_TOK) / 4;
  cvt_dist4_kernel<<<nd4 / 256, 256, 0, stream>>>((const int4*)dist, (unsigned int*)du8, nd4);

  qkv_gemm_kernel<<<dim3(N_TOK / 128, (3 * DIM_) / 32), 256, 0, stream>>>(
      xh, wh, qkv_b, qh, kh, vth);

  flash_attn_kernel<<<dim3(N_TOK / QT, NH), 128, 0, stream>>>(
      qh, kh, vth, du8, rpe, out);
}
